// Decoding_77841987272838
// MI455X (gfx1250) — compile-verified, matrix operand order checked by reading, not run on hardware
//
#include <hip/hip_runtime.h>
#include <hip/hip_bf16.h>

#define C_CELLS 512
#define G_GENES 1024
#define L_LAT 32
#define NBINS 128
#define K_H 129
#define LSTRIDE 132   /* logit row stride, padded so rows are 16B-aligned */
#define KT 9          /* k-tiles of 16 covering 144 >= 129 */
#define CT 32         /* cell tiles: 512/16 */
#define TAB_STRIDE 520 /* per-gene precompute table stride (floats) */
#define N_CUTS 1000000
#define N_FRAG 1000000
#define N_CELLS_TOTAL 100000
#define LOG_2PI 1.8378770664093453f

typedef __attribute__((ext_vector_type(16))) _Float16 v16h;
typedef __attribute__((ext_vector_type(8)))  float    v8f;

// ---------------------------------------------------------------- utilities
__device__ __forceinline__ void block_reduce_atomic(float v, float* dst) {
    __shared__ float red[256];
    red[threadIdx.x] = v;
    __syncthreads();
    for (int off = 128; off > 0; off >>= 1) {
        if ((int)threadIdx.x < off) red[threadIdx.x] += red[threadIdx.x + off];
        __syncthreads();
    }
    if (threadIdx.x == 0) atomicAdd(dst, red[0]);
}

__global__ void zero_kernel(float* __restrict__ p, int n) {
    int i = blockIdx.x * blockDim.x + threadIdx.x;
    if (i < n) p[i] = 0.0f;
}

// --------------------------------------------------- operand swizzle (f16)
__global__ void prep_a_kernel(const float* __restrict__ latent,
                              _Float16* __restrict__ Asw) {
    int idx = blockIdx.x * blockDim.x + threadIdx.x;   // CT*32*16
    if (idx >= CT * 32 * 16) return;
    int i    = idx & 15;
    int lane = (idx >> 4) & 31;
    int ct   = idx >> 9;
    int m = ct * 16 + (lane & 15);
    int k = ((lane >> 4) << 3) + (i & 7) + ((i >> 3) << 4);
    Asw[idx] = (_Float16)latent[m * L_LAT + k];
}

__global__ void prep_b_kernel(const float* __restrict__ logit_weight,
                              const int* __restrict__ genes_oi,
                              _Float16* __restrict__ Bsw) {
    int idx = blockIdx.x * blockDim.x + threadIdx.x;   // G*KT*32*16
    if (idx >= G_GENES * KT * 32 * 16) return;
    int i    = idx & 15;
    int lane = (idx >> 4) & 31;
    int gt   = idx >> 9;            // g*KT + t
    int t    = gt % KT;
    int g    = gt / KT;
    int l = ((lane >> 4) << 3) + (i & 7) + ((i >> 3) << 4);   // latent dim (WMMA K)
    int n = t * 16 + (lane & 15);                              // height index (WMMA N)
    float v = 0.0f;
    if (n < K_H) {
        int gg = genes_oi[g];
        v = logit_weight[((size_t)gg * L_LAT + l) * K_H + n];
    }
    Bsw[idx] = (_Float16)v;
}

// ------------------------------------------------------- WMMA batched GEMM
// logit[c][g][k] = sum_l latent[c][l] * lw[g][l][k]; row stride LSTRIDE.
__global__ void __launch_bounds__(256) logit_wmma_kernel(
        const _Float16* __restrict__ Asw,
        const _Float16* __restrict__ Bsw,
        float* __restrict__ logit) {
    const int lane = threadIdx.x & 31;
    const int wave = blockIdx.x * (blockDim.x >> 5) + (threadIdx.x >> 5);
    const int g  = wave >> 5;     // 0..1023
    const int ct = wave & 31;     // 0..31

    v16h a = *(const v16h*)(Asw + ((size_t)(ct * 32 + lane) << 4));
    const int n_lo   = lane & 15;
    const int m_base = ct * 16 + ((lane >> 4) << 3);

    for (int t = 0; t < KT; ++t) {
        v16h b = *(const v16h*)(Bsw + ((size_t)((g * KT + t) * 32 + lane) << 4));
        v8f c = {};
        c = __builtin_amdgcn_wmma_f32_16x16x32_f16(
                /*neg_a=*/false, a, /*neg_b=*/false, b,
                /*c_mod=*/(short)0, c, /*reuse_a=*/false, /*reuse_b=*/false);
        int n = t * 16 + n_lo;
        if (n < K_H) {
            #pragma unroll
            for (int r = 0; r < 8; ++r) {
                int cell = m_base + r;
                logit[((size_t)cell * G_GENES + g) * LSTRIDE + n] = c[r];
            }
        }
    }
}

// ------------------------------------------- per-gene spline precompute table
// Table per local gene g (stride TAB_STRIDE floats):
//   [0..128]   edges: cumulative widths with leading 0 (edges[0]=0)
//   [129..256] wbins: softmax widths (128)
//   [257..385] euh:   exp(uh) (129)
//   [386..514] coef:  0.5*(w[j-1]+w[j])*euh[j]  (w[-1]=w[128]=0)
__global__ void prep_gene_kernel(const float* __restrict__ uw_full,
                                 const float* __restrict__ uh_full,
                                 const int* __restrict__ genes_oi,
                                 float* __restrict__ gtab) {
    int g = blockIdx.x * blockDim.x + threadIdx.x;
    if (g >= G_GENES) return;
    int gg = genes_oi[g];
    const float* uw = uw_full + (size_t)gg * NBINS;
    const float* uh = uh_full + (size_t)gg * K_H;
    float* tab   = gtab + (size_t)g * TAB_STRIDE;
    float* edges = tab;
    float* wbins = tab + 129;
    float* euh   = tab + 257;
    float* coef  = tab + 386;

    float mx = -1e30f;
    for (int k = 0; k < NBINS; ++k) mx = fmaxf(mx, uw[k]);
    float s = 0.0f;
    for (int k = 0; k < NBINS; ++k) s += __expf(uw[k] - mx);
    float inv_s = __frcp_rn(s);

    float cum = 0.0f;
    edges[0] = 0.0f;
    for (int k = 0; k < NBINS; ++k) {
        float w = __expf(uw[k] - mx) * inv_s;
        wbins[k] = w;
        cum += w;
        edges[k + 1] = cum;
    }
    for (int j = 0; j < K_H; ++j) euh[j] = __expf(uh[j]);
    for (int j = 0; j < K_H; ++j) {
        float wl = (j > 0)     ? wbins[j - 1] : 0.0f;
        float wr = (j < NBINS) ? wbins[j]     : 0.0f;
        coef[j] = 0.5f * (wl + wr) * euh[j];
    }
}

// ------------------------------------------------- per-cut spline log-density
__global__ void spline_kernel(const float* __restrict__ x_all,
                              const int* __restrict__ cxg_ix,
                              const int* __restrict__ gix,
                              const float* __restrict__ gtab,
                              const float* __restrict__ logit,
                              float* __restrict__ acc) {
    int i = blockIdx.x * blockDim.x + threadIdx.x;
    float ll = 0.0f;
    if (i < N_CUTS) {
        float x = x_all[i];
        const float* tab   = gtab + (size_t)gix[i] * TAB_STRIDE;
        const float* edges = tab;
        const float* wbins = tab + 129;
        const float* euh   = tab + 257;
        const float* coef  = tab + 386;
        const float* row   = logit + (size_t)cxg_ix[i] * LSTRIDE;   // 16B-aligned

        // norm = sum_j coef[j] * exp(delta[j]) ; delta read as float4
        const float4* d4 = (const float4*)row;
        float norm = 0.0f;
        #pragma unroll 4
        for (int jj = 0; jj < 32; ++jj) {
            float4 d = d4[jj];
            int j = jj << 2;
            norm += coef[j + 0] * __expf(d.x);
            norm += coef[j + 1] * __expf(d.y);
            norm += coef[j + 2] * __expf(d.z);
            norm += coef[j + 3] * __expf(d.w);
        }
        norm += coef[128] * __expf(row[128]);

        // binary search: largest bin in [0,127] with edges[bin] <= x
        int bin = 0;
        #pragma unroll
        for (int step = 64; step >= 1; step >>= 1) {
            int nb = bin + step;
            if (nb <= 127 && x >= edges[nb]) bin = nb;
        }

        float left = edges[bin];
        float w    = wbins[bin];
        float h0   = euh[bin]     * __expf(row[bin]);
        float h1   = euh[bin + 1] * __expf(row[bin + 1]);
        float alpha = (x - left) / w;
        float dens  = (h0 + (h1 - h0) * alpha) / norm;
        ll = __logf(dens);
    }
    block_reduce_atomic(ll, acc + 0);
}

// ------------------------------------------------------- fragment segment-sum
__global__ void frag_count_kernel(const int* __restrict__ lix,
                                  float* __restrict__ counts) {
    int i = blockIdx.x * blockDim.x + threadIdx.x;
    if (i < N_FRAG) atomicAdd(counts + lix[i], 1.0f);
}

// -------------------------------------- Poisson likelihood (rho GEMV fused)
__global__ void counts_ll_kernel(const float* __restrict__ latent,
                                 const float* __restrict__ rho_weight,
                                 const float* __restrict__ counts,
                                 const int* __restrict__ genes_oi,
                                 const int* __restrict__ cells_oi,
                                 const float* __restrict__ rho_bias,
                                 const float* __restrict__ libsize,
                                 float* __restrict__ acc) {
    int idx = blockIdx.x * blockDim.x + threadIdx.x;
    float ll = 0.0f;
    if (idx < C_CELLS * G_GENES) {
        int g = idx & (G_GENES - 1);
        int c = idx >> 10;
        int gg = genes_oi[g];
        float r = 0.0f;
        #pragma unroll
        for (int l = 0; l < L_LAT; ++l)
            r = fmaf(latent[c * L_LAT + l], rho_weight[gg * L_LAT + l], r);
        float rate = rho_bias[gg] * __expf(r) * libsize[cells_oi[c]];
        float cnt = counts[idx];
        ll = cnt * __logf(rate) - rate - lgammaf(cnt + 1.0f);
    }
    block_reduce_atomic(ll, acc + 1);
}

// --------------------------------------------------------------- KL priors
__global__ void kl_lw_kernel(const float* __restrict__ logit_weight,
                             const int* __restrict__ genes_oi,
                             float* __restrict__ acc) {
    const int per_gene = L_LAT * K_H;   // 4128
    int idx = blockIdx.x * blockDim.x + threadIdx.x;
    float v = 0.0f;
    if (idx < G_GENES * per_gene) {
        int g = idx / per_gene;
        int r = idx - g * per_gene;
        float w = logit_weight[(size_t)genes_oi[g] * per_gene + r];
        v = -0.5f * w * w - 0.5f * LOG_2PI;
    }
    block_reduce_atomic(v, acc + 2);
}

__global__ void kl_rw_kernel(const float* __restrict__ rho_weight,
                             const int* __restrict__ genes_oi,
                             float* __restrict__ acc) {
    int idx = blockIdx.x * blockDim.x + threadIdx.x;
    float v = 0.0f;
    if (idx < G_GENES * L_LAT) {
        int g = idx >> 5;
        int l = idx & 31;
        float w = rho_weight[(size_t)genes_oi[g] * L_LAT + l];
        v = -0.5f * w * w - 0.5f * LOG_2PI;
    }
    block_reduce_atomic(v, acc + 3);
}

__global__ void finalize_kernel(const float* __restrict__ acc, float* __restrict__ out) {
    out[0] = (-(acc[0] + acc[1]) - acc[2] - acc[3]) / (float)N_CELLS_TOTAL;
}

// ------------------------------------------------------------------ launch
extern "C" void kernel_launch(void* const* d_in, const int* in_sizes, int n_in,
                              void* d_out, int out_size, void* d_ws, size_t ws_size,
                              hipStream_t stream) {
    const float* cut_coordinates = (const float*)d_in[0];
    const float* latent          = (const float*)d_in[1];
    const int*   genes_oi        = (const int*)  d_in[2];
    const int*   cells_oi        = (const int*)  d_in[3];
    const int*   cut_cxg_ix      = (const int*)  d_in[4];
    const int*   cut_gene_ix     = (const int*)  d_in[5];
    const int*   local_cxg_ix    = (const int*)  d_in[6];
    const float* logit_weight    = (const float*)d_in[9];
    const float* rho_weight      = (const float*)d_in[10];
    const float* unnorm_widths   = (const float*)d_in[11];
    const float* unnorm_heights  = (const float*)d_in[12];
    const float* rho_bias        = (const float*)d_in[13];
    const float* libsize         = (const float*)d_in[14];
    float* out = (float*)d_out;

    // workspace carve-up (all offsets 256B-aligned)
    char* ws = (char*)d_ws;
    const size_t logit_bytes  = (size_t)C_CELLS * G_GENES * LSTRIDE * sizeof(float);  // 276,824,064
    const size_t bsw_bytes    = (size_t)G_GENES * KT * 32 * 16 * sizeof(_Float16);    //   9,437,184
    const size_t asw_bytes    = (size_t)CT * 32 * 16 * sizeof(_Float16);              //      32,768
    const size_t counts_bytes = (size_t)C_CELLS * G_GENES * sizeof(float);            //   2,097,152
    const size_t gtab_bytes   = (size_t)G_GENES * TAB_STRIDE * sizeof(float);         //   2,129,920

    float*    ws_logit  = (float*)   (ws);
    _Float16* ws_bsw    = (_Float16*)(ws + logit_bytes);
    _Float16* ws_asw    = (_Float16*)(ws + logit_bytes + bsw_bytes);
    float*    ws_counts = (float*)   (ws + logit_bytes + bsw_bytes + asw_bytes);
    float*    ws_gtab   = (float*)   (ws + logit_bytes + bsw_bytes + asw_bytes + counts_bytes);
    float*    ws_acc    = (float*)   (ws + logit_bytes + bsw_bytes + asw_bytes + counts_bytes + gtab_bytes);

    const int B = 256;
    // 0) zero accumulators + counts (deterministic per call)
    zero_kernel<<<(C_CELLS * G_GENES + B - 1) / B, B, 0, stream>>>(ws_counts, C_CELLS * G_GENES);
    zero_kernel<<<1, B, 0, stream>>>(ws_acc, 8);

    // 1) swizzle f16 operands + per-gene spline tables
    prep_a_kernel<<<(CT * 32 * 16 + B - 1) / B, B, 0, stream>>>(latent, ws_asw);
    prep_b_kernel<<<(G_GENES * KT * 32 * 16 + B - 1) / B, B, 0, stream>>>(logit_weight, genes_oi, ws_bsw);
    prep_gene_kernel<<<(G_GENES + B - 1) / B, B, 0, stream>>>(unnorm_widths, unnorm_heights, genes_oi, ws_gtab);

    // 2) WMMA batched GEMM: 1024 genes x 32 cell-tiles, 8 waves/block
    logit_wmma_kernel<<<(G_GENES * CT) / 8, B, 0, stream>>>(ws_asw, ws_bsw, ws_logit);

    // 3) per-cut spline log-density
    spline_kernel<<<(N_CUTS + B - 1) / B, B, 0, stream>>>(
        cut_coordinates, cut_cxg_ix, cut_gene_ix, ws_gtab, ws_logit, ws_acc);

    // 4) fragment counts + Poisson likelihood (rho fused)
    frag_count_kernel<<<(N_FRAG + B - 1) / B, B, 0, stream>>>(local_cxg_ix, ws_counts);
    counts_ll_kernel<<<(C_CELLS * G_GENES + B - 1) / B, B, 0, stream>>>(
        latent, rho_weight, ws_counts, genes_oi, cells_oi, rho_bias, libsize, ws_acc);

    // 5) KL priors
    kl_lw_kernel<<<(G_GENES * L_LAT * K_H + B - 1) / B, B, 0, stream>>>(logit_weight, genes_oi, ws_acc);
    kl_rw_kernel<<<(G_GENES * L_LAT + B - 1) / B, B, 0, stream>>>(rho_weight, genes_oi, ws_acc);

    // 6) combine
    finalize_kernel<<<1, 1, 0, stream>>>(ws_acc, out);

    (void)in_sizes; (void)n_in; (void)out_size; (void)ws_size;
}
